// LanguageModel_79894981640243
// MI455X (gfx1250) — matrix operand order repaired and checked by reading, not compiled, not run on hardware
//
#include <hip/hip_runtime.h>
#include <hip/hip_bf16.h>
#include <math.h>

// ---------------------------------------------------------------------------
// Types matching the gfx1250 WMMA builtin signatures
// ---------------------------------------------------------------------------
typedef __bf16 bf16_t;
typedef bf16_t v16bf __attribute__((ext_vector_type(16)));
typedef bf16_t v8bf  __attribute__((ext_vector_type(8)));
typedef float  v8f   __attribute__((ext_vector_type(8)));
typedef int    v4i   __attribute__((ext_vector_type(4)));

#define B_  32
#define T_  64
#define D_  512
#define H_  1024
#define V_  20000
#define G4H (4 * H_)
#define BT  (B_ * T_)

// gfx1250 async LDS path detection (toolchain-dependent); fallback compiles too
#if defined(__has_builtin)
#  if __has_builtin(__builtin_amdgcn_global_load_async_to_lds_b128)
#    define HAVE_ASYNC_LDS 1
#  endif
#endif
#ifndef HAVE_ASYNC_LDS
#  define HAVE_ASYNC_LDS 0
#endif
#define AS1 __attribute__((address_space(1)))
#define AS3 __attribute__((address_space(3)))

// fp32 -> bf16 round-to-nearest-even (bit-level; no cvt dependency)
__device__ __forceinline__ bf16_t f2bf(float f) {
  union { float f; unsigned u; } v; v.f = f;
  unsigned r = v.u + 0x7FFFu + ((v.u >> 16) & 1u);
  union { unsigned short s; bf16_t b; } o; o.s = (unsigned short)(r >> 16);
  return o.b;
}

__device__ __forceinline__ v8f wmma_bf16(v16bf a, v16bf b, v8f c) {
  // (neg_a, A, neg_b, B, c_mod, C, reuse_a, reuse_b)
  return __builtin_amdgcn_wmma_f32_16x16x32_bf16(false, a, false, b, (short)0, c,
                                                 false, false);
}

// A fragment: 16x32 bf16, row-major A (lda in elements).
// ISA layout: lane<16 -> row M=lane, K = kb+{0..7, 16..23}; lane>=16 -> K = kb+{8..15, 24..31}
__device__ __forceinline__ v16bf load_a_frag(const bf16_t* __restrict__ A,
                                             int row0, long lda, int kb, int lane) {
  int m  = row0 + (lane & 15);
  int kh = (lane >> 4) & 1;
  const bf16_t* p = A + (size_t)m * lda + kb + kh * 8;
  v8bf lo = *(const v8bf*)(p);        // K = kb + kh*8 + 0..7
  v8bf hi = *(const v8bf*)(p + 16);   // K = kb + 16 + kh*8 + 0..7
  return __builtin_shufflevector(lo, hi, 0,1,2,3,4,5,6,7,8,9,10,11,12,13,14,15);
}

// B fragment: logical 32x16, from row-major W[N,K] (we compute A @ W^T).
// lane<16 -> col N=lane, K = kb+{0..7, 16..23}; lane>=16 -> K = kb+{8..15, 24..31}
__device__ __forceinline__ v16bf load_b_frag(const bf16_t* __restrict__ W,
                                             int col0, long ldw, int kb, int lane) {
  int n  = col0 + (lane & 15);
  int kh = (lane >> 4) & 1;
  const bf16_t* p = W + (size_t)n * ldw + kb + kh * 8;
  v8bf lo = *(const v8bf*)(p);
  v8bf hi = *(const v8bf*)(p + 16);
  return __builtin_shufflevector(lo, hi, 0,1,2,3,4,5,6,7,8,9,10,11,12,13,14,15);
}

// ---------------------------------------------------------------------------
// fp32 -> bf16 bulk convert
// ---------------------------------------------------------------------------
__global__ void __launch_bounds__(256)
k_f32_to_bf16(const float* __restrict__ in, bf16_t* __restrict__ out, long n) {
  long i = (long)blockIdx.x * 256 + threadIdx.x;
  if (i < n) out[i] = f2bf(in[i]);
}

// init: h0 -> bf16, c = 0, biasSum = b_ih + b_hh
__global__ void __launch_bounds__(256)
k_init(const float* __restrict__ hidden, const float* __restrict__ b_ih,
       const float* __restrict__ b_hh, bf16_t* __restrict__ h0_bf,
       float* __restrict__ c, float* __restrict__ biasSum) {
  int i = blockIdx.x * 256 + threadIdx.x;
  if (i < B_ * H_) { h0_bf[i] = f2bf(hidden[i]); c[i] = 0.0f; }
  if (i < G4H)     biasSum[i] = b_ih[i] + b_hh[i];
}

// ---------------------------------------------------------------------------
// Generic GEMM: C[M,N] = A_bf[M,K] @ W_bf[N,K]^T + bias[N]
// One wave computes a 32x32 tile (2x2 WMMA register tiles). Grid exact.
// ---------------------------------------------------------------------------
__global__ void __launch_bounds__(256)
k_gemm_bf16(const bf16_t* __restrict__ A, int lda,
            const bf16_t* __restrict__ W, int ldw,
            const float* __restrict__ bias,
            float* __restrict__ C, int ldc,
            int Nt2, int K, int njobs) {
  const int lane = threadIdx.x & 31;
  const int wave = threadIdx.x >> 5;
  const int job  = blockIdx.x * 8 + wave;
  if (job >= njobs) return;                 // wave-uniform -> EXEC all-1s for WMMA
  const int m2 = job / Nt2, n2 = job % Nt2;
  const int row0 = m2 * 32, col0 = n2 * 32;

  v8f acc00 = {}, acc01 = {}, acc10 = {}, acc11 = {};
  for (int k = 0; k < K; k += 32) {
    v16bf a0 = load_a_frag(A, row0,      lda, k, lane);
    v16bf a1 = load_a_frag(A, row0 + 16, lda, k, lane);
    v16bf b0 = load_b_frag(W, col0,      ldw, k, lane);
    v16bf b1 = load_b_frag(W, col0 + 16, ldw, k, lane);
    acc00 = wmma_bf16(a0, b0, acc00);
    acc01 = wmma_bf16(a0, b1, acc01);
    acc10 = wmma_bf16(a1, b0, acc10);
    acc11 = wmma_bf16(a1, b1, acc11);
  }

  const int nlo = lane & 15;
  const int mh  = (lane >> 4) * 8;
  const float bz0 = bias[col0 + nlo], bz1 = bias[col0 + 16 + nlo];
#pragma unroll
  for (int r = 0; r < 8; ++r) {
    const size_t mA = (size_t)(row0 + mh + r) * ldc;
    const size_t mB = (size_t)(row0 + 16 + mh + r) * ldc;
    C[mA + col0 + nlo]      = acc00[r] + bz0;
    C[mA + col0 + 16 + nlo] = acc01[r] + bz1;
    C[mB + col0 + nlo]      = acc10[r] + bz0;
    C[mB + col0 + 16 + nlo] = acc11[r] + bz1;
  }
}

// ---------------------------------------------------------------------------
// Logits GEMM with LDS-staged A (gfx1250 async-to-LDS path):
// C[BT,V] = Hall[BT,H] @ Wlin[V,H]^T + b_lin.
// One block = one 32-row M-tile shared by 8 waves, each wave a 32-col N-tile.
// A strip (32 x 1024 bf16 = 64 KB) staged once per block via
// global_load_async_to_lds_b128 (+ s_wait_asynccnt), then read via ds loads.
// ---------------------------------------------------------------------------
__global__ void __launch_bounds__(256)
k_gemm_logits(const bf16_t* __restrict__ A,    // Hall [BT, H]
              const bf16_t* __restrict__ W,    // Wlin_bf [V, H]
              const float* __restrict__ bias,  // b_lin [V]
              float* __restrict__ C,           // out [BT, V]
              int Ngroups) {
  __shared__ bf16_t As[32 * H_];               // 64 KB

  const int lane = threadIdx.x & 31;
  const int wave = threadIdx.x >> 5;
  const int m2 = blockIdx.x / Ngroups;
  const int ng = blockIdx.x % Ngroups;
  const int row0 = m2 * 32;

  // ---- Stage A(32 x H) into LDS: 4096 x 16B chunks, 16 per thread ----
  const bf16_t* gA = A + (size_t)row0 * H_;    // 64 KB contiguous strip
  for (int cidx = threadIdx.x; cidx < 32 * (H_ / 8); cidx += 256) {
    const int r  = cidx >> 7;                  // row in strip (128 chunks/row)
    const int ch = (cidx & 127) * 8;           // element offset within row
#if HAVE_ASYNC_LDS
    __builtin_amdgcn_global_load_async_to_lds_b128(
        (AS1 v4i*)(void*)(gA + (size_t)r * H_ + ch),
        (AS3 v4i*)(void*)&As[r * H_ + ch], 0, 0);
#else
    *(v8bf*)&As[r * H_ + ch] = *(const v8bf*)(gA + (size_t)r * H_ + ch);
#endif
  }
#if HAVE_ASYNC_LDS
#  if __has_builtin(__builtin_amdgcn_s_wait_asynccnt)
  __builtin_amdgcn_s_wait_asynccnt(0);
#  else
  asm volatile("s_wait_asynccnt 0" ::: "memory");
#  endif
#endif
  __syncthreads();

  const int n2 = ng * 8 + wave;                // wave-uniform
  if (n2 < V_ / 32) {
    const int col0 = n2 * 32;
    const int mlo = lane & 15;
    const int kh8 = ((lane >> 4) & 1) * 8;

    v8f acc00 = {}, acc01 = {}, acc10 = {}, acc11 = {};
    for (int k = 0; k < H_; k += 32) {
      // A fragments from LDS (documented 16-bit A per-lane layout)
      const bf16_t* p0 = &As[(unsigned)(mlo * H_ + k + kh8)];
      const bf16_t* p1 = &As[(unsigned)((16 + mlo) * H_ + k + kh8)];
      v8bf a0lo = *(const v8bf*)p0, a0hi = *(const v8bf*)(p0 + 16);
      v8bf a1lo = *(const v8bf*)p1, a1hi = *(const v8bf*)(p1 + 16);
      v16bf a0 = __builtin_shufflevector(a0lo, a0hi, 0,1,2,3,4,5,6,7,8,9,10,11,12,13,14,15);
      v16bf a1 = __builtin_shufflevector(a1lo, a1hi, 0,1,2,3,4,5,6,7,8,9,10,11,12,13,14,15);
      v16bf b0 = load_b_frag(W, col0,      H_, k, lane);
      v16bf b1 = load_b_frag(W, col0 + 16, H_, k, lane);
      acc00 = wmma_bf16(a0, b0, acc00);
      acc01 = wmma_bf16(a0, b1, acc01);
      acc10 = wmma_bf16(a1, b0, acc10);
      acc11 = wmma_bf16(a1, b1, acc11);
    }

    const int nlo = lane & 15;
    const int mh  = (lane >> 4) * 8;
    const float bz0 = bias[col0 + nlo], bz1 = bias[col0 + 16 + nlo];
#pragma unroll
    for (int r = 0; r < 8; ++r) {
      const size_t mA = (size_t)(row0 + mh + r) * V_;
      const size_t mB = (size_t)(row0 + 16 + mh + r) * V_;
      C[mA + col0 + nlo]      = acc00[r] + bz0;
      C[mA + col0 + 16 + nlo] = acc01[r] + bz1;
      C[mB + col0 + nlo]      = acc10[r] + bz0;
      C[mB + col0 + 16 + nlo] = acc11[r] + bz1;
    }
  }
}

// ---------------------------------------------------------------------------
// One LSTM timestep: gates[32,4H] = Xproj[:,t,:] + h_{t-1} @ W_hh^T, fused cell.
// Grid: H/16 = 64 blocks x 256 threads (8 waves).
// ---------------------------------------------------------------------------
__global__ void __launch_bounds__(256)
k_lstm_step(const bf16_t* __restrict__ Aprev, long lda,   // h_{t-1} bf16
            const bf16_t* __restrict__ Whh,               // [4H, H] bf16
            const float* __restrict__ Xproj,              // [B*T, 4H] f32
            float* __restrict__ c,                        // [B, H] f32 state
            bf16_t* __restrict__ Hall,                    // [B*T, H] bf16
            int t) {
  __shared__ float gsm[4][32][17];

  const int lane = threadIdx.x & 31;
  const int wave = threadIdx.x >> 5;
  const int sect = wave >> 1;                  // 0..3 : i,f,g,o
  const int mt   = (wave & 1) * 16;            // row tile
  const int col_local = blockIdx.x * 16;       // column within H
  const int col_abs   = sect * H_ + col_local; // column within 4H

  v8f acc = {};
  for (int k = 0; k < H_; k += 32) {
    v16bf a = load_a_frag(Aprev, mt, lda, k, lane);
    v16bf b = load_b_frag(Whh, col_abs, H_, k, lane);
    acc = wmma_bf16(a, b, acc);
  }

  const int nlo = lane & 15;
  const int mh  = (lane >> 4) * 8;
#pragma unroll
  for (int r = 0; r < 8; ++r) {
    const int m = mt + mh + r;                 // batch row 0..31
    gsm[sect][m][nlo] =
        acc[r] + Xproj[((long)m * T_ + t) * G4H + col_abs + nlo];
  }
  __syncthreads();

  for (int e = threadIdx.x; e < 32 * 16; e += 256) {
    const int b = e >> 4, j = e & 15;
    const int hcol = col_local + j;
    const float gi = gsm[0][b][j], gf = gsm[1][b][j];
    const float gg = gsm[2][b][j], go = gsm[3][b][j];
    const float i_ = 1.0f / (1.0f + expf(-gi));
    const float f_ = 1.0f / (1.0f + expf(-gf));
    const float g_ = tanhf(gg);
    const float o_ = 1.0f / (1.0f + expf(-go));
    const float cv = f_ * c[b * H_ + hcol] + i_ * g_;
    c[b * H_ + hcol] = cv;
    Hall[((long)b * T_ + t) * H_ + hcol] = f2bf(o_ * tanhf(cv));
  }
}

// ---------------------------------------------------------------------------
// In-place row-wise log-softmax over V=20000; one block per row.
// ---------------------------------------------------------------------------
__global__ void __launch_bounds__(256)
k_logsoftmax(float* __restrict__ out, int V) {
  float* p = out + (long)blockIdx.x * V;
  __shared__ float smax[8], ssum[8];
  const int lane = threadIdx.x & 31, wv = threadIdx.x >> 5;

  float mx = -3.4e38f;
  for (int i = threadIdx.x; i < V; i += 256) mx = fmaxf(mx, p[i]);
#pragma unroll
  for (int o = 16; o > 0; o >>= 1) mx = fmaxf(mx, __shfl_down(mx, o, 32));
  if (lane == 0) smax[wv] = mx;
  __syncthreads();
  mx = smax[0];
#pragma unroll
  for (int i = 1; i < 8; ++i) mx = fmaxf(mx, smax[i]);

  float s = 0.0f;
  for (int i = threadIdx.x; i < V; i += 256) s += expf(p[i] - mx);
#pragma unroll
  for (int o = 16; o > 0; o >>= 1) s += __shfl_down(s, o, 32);
  if (lane == 0) ssum[wv] = s;
  __syncthreads();
  s = 0.0f;
#pragma unroll
  for (int i = 0; i < 8; ++i) s += ssum[i];

  const float lse = mx + logf(s);
  for (int i = threadIdx.x; i < V; i += 256) p[i] -= lse;
}

// ---------------------------------------------------------------------------
// Host orchestration
// ---------------------------------------------------------------------------
extern "C" void kernel_launch(void* const* d_in, const int* in_sizes, int n_in,
                              void* d_out, int out_size, void* d_ws, size_t ws_size,
                              hipStream_t stream) {
  (void)in_sizes; (void)n_in; (void)out_size; (void)ws_size;

  const float* iseq   = (const float*)d_in[0];   // [B,T,D]
  const float* h0     = (const float*)d_in[1];   // [B,H]
  const float* W_ih   = (const float*)d_in[2];   // [4H,D]
  const float* W_hh   = (const float*)d_in[3];   // [4H,H]
  const float* b_ih   = (const float*)d_in[4];   // [4H]
  const float* b_hh   = (const float*)d_in[5];   // [4H]
  const float* W_lin  = (const float*)d_in[6];   // [V,H]
  const float* b_lin  = (const float*)d_in[7];   // [V]
  float* out = (float*)d_out;                    // [B,T,V]

  // Workspace carve-up (all sizes multiples of 256B)
  char* ws = (char*)d_ws;
  bf16_t* Wih_bf  = (bf16_t*)ws;                      ws += (size_t)G4H * D_ * 2;
  bf16_t* Whh_bf  = (bf16_t*)ws;                      ws += (size_t)G4H * H_ * 2;
  bf16_t* Wlin_bf = (bf16_t*)ws;                      ws += (size_t)V_  * H_ * 2;
  bf16_t* X_bf    = (bf16_t*)ws;                      ws += (size_t)BT  * D_ * 2;
  bf16_t* Hall    = (bf16_t*)ws;                      ws += (size_t)BT  * H_ * 2;
  bf16_t* h0_bf   = (bf16_t*)ws;                      ws += (size_t)B_  * H_ * 2;
  float*  Xproj   = (float*)ws;                       ws += (size_t)BT  * G4H * 4;
  float*  cstate  = (float*)ws;                       ws += (size_t)B_  * H_ * 4;
  float*  biasSum = (float*)ws;                       ws += (size_t)G4H * 4;

  auto cvt = [&](const float* src, bf16_t* dst, long n) {
    k_f32_to_bf16<<<(unsigned)((n + 255) / 256), 256, 0, stream>>>(src, dst, n);
  };
  cvt(W_ih,  Wih_bf,  (long)G4H * D_);
  cvt(W_hh,  Whh_bf,  (long)G4H * H_);
  cvt(W_lin, Wlin_bf, (long)V_  * H_);
  cvt(iseq,  X_bf,    (long)BT  * D_);

  k_init<<<(B_ * H_ + 255) / 256, 256, 0, stream>>>(h0, b_ih, b_hh, h0_bf,
                                                    cstate, biasSum);

  // Xproj[BT,4H] = X @ W_ih^T + (b_ih+b_hh)
  {
    const int Nt2 = G4H / 32, njobs = (BT / 32) * Nt2;
    k_gemm_bf16<<<njobs / 8, 256, 0, stream>>>(X_bf, D_, Wih_bf, D_, biasSum,
                                               Xproj, G4H, Nt2, D_, njobs);
  }

  // Recurrence: 64 dependent fused gate-GEMM + cell kernels
  for (int t = 0; t < T_; ++t) {
    const bf16_t* Aprev = (t == 0) ? h0_bf : (Hall + (size_t)(t - 1) * H_);
    const long lda = (t == 0) ? (long)H_ : (long)T_ * H_;
    k_lstm_step<<<H_ / 16, 256, 0, stream>>>(Aprev, lda, Whh_bf, Xproj, cstate,
                                             Hall, t);
  }

  // Logits: out[BT,V] = Hall @ W_lin^T + b_lin, LDS-staged A strip per block
  {
    const int Nt2 = V_ / 32;                 // 625
    const int Ngroups = (Nt2 + 7) / 8;       // 79
    k_gemm_logits<<<(BT / 32) * Ngroups, 256, 0, stream>>>(Hall, Wlin_bf, b_lin,
                                                           out, Ngroups);
  }

  // In-place log-softmax over V per (b,t) row
  k_logsoftmax<<<BT, 256, 0, stream>>>(out, V_);
}
